// AttentionWithRoPE_71751723647808
// MI455X (gfx1250) — matrix operand order repaired and checked by reading, compiled, not run on hardware
//
#include <hip/hip_runtime.h>

// ---------------------------------------------------------------------------
// Types for CDNA5 WMMA / TDM (wave32)
// ---------------------------------------------------------------------------
typedef __attribute__((ext_vector_type(16))) __bf16        v16bf;
typedef __attribute__((ext_vector_type(8)))  float         v8f;
typedef __attribute__((ext_vector_type(4)))  unsigned int  u32x4;
typedef __attribute__((ext_vector_type(8)))  int           i32x8;
typedef __attribute__((ext_vector_type(4)))  int           i32x4;

#define D_MODEL 1024
#define NHEAD   16
#define DKH     64
#define BATCH   2
#define SEQ     2048
#define TOKENS  (BATCH * SEQ)          // 4096
#define HEAD_ELEMS ((size_t)BATCH * NHEAD * SEQ * DKH)  // 4,194,304 per buffer

// ---------------------------------------------------------------------------
// Helpers
// ---------------------------------------------------------------------------
__device__ __forceinline__ unsigned short f32_to_bf16(float x) {
    unsigned int u = __float_as_uint(x);
    unsigned int r = u + 0x7FFFu + ((u >> 16) & 1u);   // round-to-nearest-even
    return (unsigned short)(r >> 16);
}
__device__ __forceinline__ float bf16_to_f32(unsigned short h) {
    return __uint_as_float(((unsigned int)h) << 16);
}

__device__ __forceinline__ v8f wmma_bf16(v16bf a, v16bf b, v8f c) {
    // 8 args: (neg_a, A, neg_b, B, c_mod, C, reuse_a, reuse_b)
    return __builtin_amdgcn_wmma_f32_16x16x32_bf16(false, a, false, b,
                                                   (short)0, c, false, false);
}

// A-matrix fragment (16x32 bf16), ISA 7.12.2 layout.
__device__ __forceinline__ v16bf load_frag_a(const unsigned short* base, int row,
                                             int koff, int lane, int stride) {
    const int l  = lane & 15;
    const int hi = lane >> 4;
    const unsigned short* p = base + (size_t)(row + l) * stride + koff + hi * 8;
    union { v16bf v; u32x4 u[2]; } f;
    f.u[0] = *(const u32x4*)(p);        // K: koff + hi*8 + 0..7
    f.u[1] = *(const u32x4*)(p + 16);   // K: koff + hi*8 + 16..23
    return f.v;
}

// B-matrix fragment (32x16 bf16); source stored K-contiguous per column [N][K].
__device__ __forceinline__ v16bf load_frag_b(const unsigned short* base, int n0,
                                             int koff, int lane, int stride) {
    const int l  = lane & 15;
    const int hi = lane >> 4;
    const unsigned short* p = base + (size_t)(n0 + l) * stride + koff + hi * 8;
    union { v16bf v; u32x4 u[2]; } f;
    f.u[0] = *(const u32x4*)(p);        // K: koff + hi*8 + 0..7
    f.u[1] = *(const u32x4*)(p + 16);   // K: koff + hi*8 + 16..23
    return f.v;
}

// ---------------------------------------------------------------------------
// Tensor Data Mover: 2-D tile load (global -> LDS), bf16 elements.
// D# built per ISA cdna5 §8.3/8.4: group0 {count=1, lds_addr, global_addr,
// type=2}, group1 {data_size=2B, tensor/tile dims, dim0 stride}.
// Toolchain declares the 6-arg builtin: (g0, g1, g2, g3, g4, cpol).
// ---------------------------------------------------------------------------
__device__ __forceinline__ void tdm_load_2d_bf16(unsigned int lds_addr,
                                                 const void* gaddr,
                                                 unsigned int dim0,        // row length (elems)
                                                 unsigned int dim1,        // #rows
                                                 unsigned int tile0,       // tile width (elems)
                                                 unsigned int tile1,       // tile height
                                                 unsigned int stride0)     // row stride (elems)
{
    unsigned long long ga = (unsigned long long)(size_t)gaddr;
    u32x4 g0;
    g0[0] = 1u;                                        // count=1, user descriptor
    g0[1] = lds_addr;                                  // lds_addr        [63:32]
    g0[2] = (unsigned int)(ga & 0xFFFFFFFFu);          // global_addr[31:0]
    g0[3] = (unsigned int)((ga >> 32) & 0x01FFFFFFu)   // global_addr[56:32]
          | (2u << 30);                                // type=2 ("image")
    i32x8 g1;
    g1[0] = (int)(1u << 16);                           // wg_mask=0, data_size=1 (2 bytes)
    g1[1] = (int)((dim0 & 0xFFFFu) << 16);             // tensor_dim0[15:0]  @ bits 63:48
    g1[2] = (int)((dim0 >> 16) | ((dim1 & 0xFFFFu) << 16)); // dim0 hi | dim1 lo
    g1[3] = (int)((dim1 >> 16) | (tile0 << 16));       // dim1 hi | tile_dim0
    g1[4] = (int)(tile1);                              // tile_dim1 (tile_dim2 = 0)
    g1[5] = (int)(stride0);                            // tensor_dim0_stride[31:0]
    g1[6] = 0;
    g1[7] = 0;
    i32x4 z4 = {0, 0, 0, 0};
    i32x8 z8 = {0, 0, 0, 0, 0, 0, 0, 0};
    __builtin_amdgcn_tensor_load_to_lds(g0, g1, z4, z4, z8, 0);
}

// Async global -> LDS 128-bit copy (ASYNCcnt path), ISA §15.18.3 opcode 98.
__device__ __forceinline__ void async_copy_b128(unsigned int lds_off, const void* g) {
    asm volatile("global_load_async_to_lds_b128 %0, %1, off"
                 :: "v"(lds_off), "v"(g) : "memory");
}
__device__ __forceinline__ void wait_asynccnt0() {
    asm volatile("s_wait_asynccnt 0x0" ::: "memory");
}

// ---------------------------------------------------------------------------
// Kernel 1: QKV projection GEMM.  C = X(f32)[4096,1024] * W(f32)[1024,1024]+b
// Output written as bf16 into (b, h, s, dk) layout for the attention kernel.
// Block tile 64(M) x 128(N), 256 threads = 8 waves (2x4), each wave 32x32.
// ---------------------------------------------------------------------------
__global__ __launch_bounds__(256) void gemm_qkv_kernel(
    const float* __restrict__ A, const float* __restrict__ W,
    const float* __restrict__ bias, unsigned short* __restrict__ dst)
{
    __shared__ unsigned short sA[64 * 32];    // [M][K]
    __shared__ unsigned short sB[128 * 32];   // [N][K]  (transposed during stage)

    const int tid  = threadIdx.x;
    const int lane = tid & 31;
    const int w    = tid >> 5;
    const int wrow = w >> 2;
    const int wcol = w & 3;
    const int m0   = blockIdx.y * 64;
    const int n0   = blockIdx.x * 128;

    v8f acc[2][2] = {};

    for (int k0 = 0; k0 < D_MODEL; k0 += 32) {
        if (k0 + 32 < D_MODEL)   // pull next W slice toward GL2
            __builtin_prefetch(W + (size_t)(k0 + 32 + (tid >> 3)) * D_MODEL + n0, 0, 0);

        // stage A: 64x32 f32 -> bf16
        #pragma unroll
        for (int t = 0; t < 2; ++t) {
            int i   = tid + 256 * t;
            int row = i >> 3;
            int kq  = (i & 7) << 2;
            float4 v = *(const float4*)(A + (size_t)(m0 + row) * D_MODEL + k0 + kq);
            uint2 pp;
            pp.x = (unsigned int)f32_to_bf16(v.x) | ((unsigned int)f32_to_bf16(v.y) << 16);
            pp.y = (unsigned int)f32_to_bf16(v.z) | ((unsigned int)f32_to_bf16(v.w) << 16);
            *(uint2*)&sA[row * 32 + kq] = pp;
        }
        // stage B: 32x128 f32 -> bf16, transposed to [N][K]
        #pragma unroll
        for (int t = 0; t < 4; ++t) {
            int i  = tid + 256 * t;
            int kk = i >> 5;
            int nq = (i & 31) << 2;
            float4 v = *(const float4*)(W + (size_t)(k0 + kk) * D_MODEL + n0 + nq);
            sB[(nq + 0) * 32 + kk] = f32_to_bf16(v.x);
            sB[(nq + 1) * 32 + kk] = f32_to_bf16(v.y);
            sB[(nq + 2) * 32 + kk] = f32_to_bf16(v.z);
            sB[(nq + 3) * 32 + kk] = f32_to_bf16(v.w);
        }
        __syncthreads();

        v16bf a0 = load_frag_a(sA, wrow * 32,      0, lane, 32);
        v16bf a1 = load_frag_a(sA, wrow * 32 + 16, 0, lane, 32);
        v16bf b0 = load_frag_b(sB, wcol * 32,      0, lane, 32);
        v16bf b1 = load_frag_b(sB, wcol * 32 + 16, 0, lane, 32);
        acc[0][0] = wmma_bf16(a0, b0, acc[0][0]);
        acc[0][1] = wmma_bf16(a0, b1, acc[0][1]);
        acc[1][0] = wmma_bf16(a1, b0, acc[1][0]);
        acc[1][1] = wmma_bf16(a1, b1, acc[1][1]);
        __syncthreads();
    }

    const int l  = lane & 15;
    const int hi = lane >> 4;
    #pragma unroll
    for (int i = 0; i < 2; ++i) {
        #pragma unroll
        for (int j = 0; j < 2; ++j) {
            int n = n0 + wcol * 32 + j * 16 + l;
            float bn = bias[n];
            int h = n >> 6, dd = n & 63;
            #pragma unroll
            for (int r = 0; r < 8; ++r) {
                int m = m0 + wrow * 32 + i * 16 + r + hi * 8;
                int b = m >> 11, s = m & 2047;
                float v = acc[i][j][r] + bn;
                dst[(((size_t)(b * NHEAD + h) * SEQ + s) * DKH) + dd] = f32_to_bf16(v);
            }
        }
    }
}

// ---------------------------------------------------------------------------
// Kernel 2: RoPE applied in-place to q and k (bf16), layout (b,h,s,dk).
// ---------------------------------------------------------------------------
__global__ __launch_bounds__(256) void rope_kernel(unsigned short* __restrict__ q,
                                                   unsigned short* __restrict__ k)
{
    const int idx = blockIdx.x * blockDim.x + threadIdx.x;   // B*H*S*32 = 2^21
    const int j   = idx & 31;
    const int s   = (idx >> 5) & (SEQ - 1);
    const int bh  = idx >> 16;
    const size_t base = ((size_t)bh * SEQ + s) * DKH;

    const float inv = exp2f(-(float)j * 0.41524101186092029f); // log2(1e4)/32
    const float th  = (float)s * inv;
    float sn, cs;
    __sincosf(th, &sn, &cs);

    float q0 = bf16_to_f32(q[base + j]);
    float q1 = bf16_to_f32(q[base + j + 32]);
    q[base + j]      = f32_to_bf16(q0 * cs - q1 * sn);
    q[base + j + 32] = f32_to_bf16(q1 * cs + q0 * sn);

    float k0 = bf16_to_f32(k[base + j]);
    float k1 = bf16_to_f32(k[base + j + 32]);
    k[base + j]      = f32_to_bf16(k0 * cs - k1 * sn);
    k[base + j + 32] = f32_to_bf16(k1 * cs + k0 * sn);
}

// ---------------------------------------------------------------------------
// Kernel 3: flash attention (non-causal, online softmax), bf16 WMMA.
// K/V tiles staged by the Tensor Data Mover (wave 0 issues the DMA), V then
// transposed LDS->LDS for the PV B-fragments.
// grid = (S/64, H, B); block = 128 threads = 4 waves; wave owns 16 q-rows.
// ---------------------------------------------------------------------------
__global__ __launch_bounds__(128) void flash_attn_kernel(
    const unsigned short* __restrict__ Q, const unsigned short* __restrict__ K,
    const unsigned short* __restrict__ V, unsigned short* __restrict__ O)
{
    __shared__ unsigned short sK [64 * 64];       // [key][d]  (TDM dest)
    __shared__ unsigned short sV [64 * 64];       // [key][d]  (TDM dest)
    __shared__ unsigned short sVt[64 * 64];       // [d][key]
    __shared__ unsigned short sP [4 * 16 * 64];   // per-wave P tile

    const int tid  = threadIdx.x;
    const int lane = tid & 31;
    const int w    = tid >> 5;
    const int l    = lane & 15;
    const int hi   = lane >> 4;
    const int bh   = blockIdx.z * NHEAD + blockIdx.y;

    const unsigned short* Qh = Q + (size_t)bh * SEQ * DKH;
    const unsigned short* Kh = K + (size_t)bh * SEQ * DKH;
    const unsigned short* Vh = V + (size_t)bh * SEQ * DKH;

    const unsigned int ldsK = (unsigned int)(size_t)(&sK[0]);
    const unsigned int ldsV = (unsigned int)(size_t)(&sV[0]);

    // Q fragments for this wave's 16 rows (dk=64 => two A frags), kept resident
    const int qrow = blockIdx.x * 64 + w * 16;
    v16bf aq[2];
    #pragma unroll
    for (int t = 0; t < 2; ++t)
        aq[t] = load_frag_a(Qh, qrow, t * 32, lane, DKH);

    float m_r[8], l_r[8];
    #pragma unroll
    for (int r = 0; r < 8; ++r) { m_r[r] = -3.0e38f; l_r[r] = 0.0f; }
    v8f o[4] = {};

    unsigned short* sPw = sP + w * (16 * 64);

    for (int kb = 0; kb < SEQ / 64; ++kb) {
        const int kb0 = kb * 64;

        // --- TDM: DMA 64x64 bf16 K and V tiles into LDS (wave 0 issues) ---
        if (w == 0) {
            tdm_load_2d_bf16(ldsK, Kh + (size_t)kb0 * DKH, 64, 64, 64, 64, 64);
            tdm_load_2d_bf16(ldsV, Vh + (size_t)kb0 * DKH, 64, 64, 64, 64, 64);
            __builtin_amdgcn_s_wait_tensorcnt(0);
        }
        __syncthreads();

        // transpose V tile in LDS: sVt[d][key] = sV[key][d]
        #pragma unroll
        for (int t = 0; t < 16; ++t) {
            int u   = tid + 128 * t;           // 2048 dwords
            int row = u >> 5;                  // key
            int dp  = (u & 31) << 1;           // d pair
            unsigned int val = *(const unsigned int*)(sV + row * 64 + dp);
            sVt[(dp + 0) * 64 + row] = (unsigned short)(val & 0xFFFFu);
            sVt[(dp + 1) * 64 + row] = (unsigned short)(val >> 16);
        }
        __syncthreads();

        // S = Q * K^T for 4 key tiles of 16
        v8f st[4];
        #pragma unroll
        for (int kt = 0; kt < 4; ++kt) {
            v8f s = {};
            #pragma unroll
            for (int t = 0; t < 2; ++t) {
                v16bf bk = load_frag_b(sK, kt * 16, t * 32, lane, DKH);
                s = wmma_bf16(aq[t], bk, s);
            }
            st[kt] = s * 0.125f;               // 1/sqrt(dk)
        }

        // online softmax: row stats live across 16 lanes of the same half-wave
        #pragma unroll
        for (int r = 0; r < 8; ++r) {
            float mx = fmaxf(fmaxf(st[0][r], st[1][r]), fmaxf(st[2][r], st[3][r]));
            #pragma unroll
            for (int msk = 1; msk < 16; msk <<= 1)
                mx = fmaxf(mx, __shfl_xor(mx, msk, 32));
            float mn   = fmaxf(m_r[r], mx);
            float corr = __expf(m_r[r] - mn);
            m_r[r] = mn;
            l_r[r] *= corr;
            #pragma unroll
            for (int j = 0; j < 4; ++j) o[j][r] = o[j][r] * corr;
            float rs = 0.0f;
            #pragma unroll
            for (int kt = 0; kt < 4; ++kt) {
                float p = __expf(st[kt][r] - mn);
                st[kt][r] = p;
                rs += p;
            }
            #pragma unroll
            for (int msk = 1; msk < 16; msk <<= 1)
                rs += __shfl_xor(rs, msk, 32);
            l_r[r] += rs;
        }

        // spill P (C layout) to LDS so it can be reloaded as an A fragment
        #pragma unroll
        for (int kt = 0; kt < 4; ++kt)
            #pragma unroll
            for (int r = 0; r < 8; ++r)
                sPw[(r + hi * 8) * 64 + kt * 16 + l] = f32_to_bf16(st[kt][r]);
        __syncthreads();

        // O += P * V
        v16bf pa0 = load_frag_a(sPw, 0, 0,  lane, 64);
        v16bf pa1 = load_frag_a(sPw, 0, 32, lane, 64);
        #pragma unroll
        for (int j = 0; j < 4; ++j) {
            v16bf vb0 = load_frag_b(sVt, j * 16, 0,  lane, 64);
            v16bf vb1 = load_frag_b(sVt, j * 16, 32, lane, 64);
            o[j] = wmma_bf16(pa0, vb0, o[j]);
            o[j] = wmma_bf16(pa1, vb1, o[j]);
        }
        __syncthreads();
    }

    // normalize and write out as (b, s, h, dk) == row-major [token][D]
    #pragma unroll
    for (int r = 0; r < 8; ++r) {
        float inv = 1.0f / l_r[r];
        int srow = blockIdx.x * 64 + w * 16 + r + hi * 8;
        size_t base = (((size_t)blockIdx.z * SEQ + srow) * NHEAD + blockIdx.y) * DKH;
        #pragma unroll
        for (int j = 0; j < 4; ++j)
            O[base + j * 16 + l] = f32_to_bf16(o[j][r] * inv);
    }
}

// ---------------------------------------------------------------------------
// Kernel 4: output projection.  attn(bf16)[4096,1024] * Wo(f32) + bo -> f32
// A tile staged via async global->LDS (ASYNCcnt path, no VGPR round-trip).
// ---------------------------------------------------------------------------
__global__ __launch_bounds__(256) void gemm_out_kernel(
    const unsigned short* __restrict__ A, const float* __restrict__ W,
    const float* __restrict__ bias, float* __restrict__ out)
{
    __shared__ unsigned short sA[64 * 32];
    __shared__ unsigned short sB[128 * 32];

    const int tid  = threadIdx.x;
    const int lane = tid & 31;
    const int w    = tid >> 5;
    const int wrow = w >> 2;
    const int wcol = w & 3;
    const int m0   = blockIdx.y * 64;
    const int n0   = blockIdx.x * 128;

    v8f acc[2][2] = {};

    for (int k0 = 0; k0 < D_MODEL; k0 += 32) {
        if (k0 + 32 < D_MODEL)
            __builtin_prefetch(W + (size_t)(k0 + 32 + (tid >> 3)) * D_MODEL + n0, 0, 0);

        // stage A (already bf16): async DMA one 16B granule per thread
        {
            int row = tid >> 2;
            int kq  = (tid & 3) << 3;
            unsigned int lds_off = (unsigned int)(size_t)(&sA[row * 32 + kq]);
            async_copy_b128(lds_off, A + (size_t)(m0 + row) * D_MODEL + k0 + kq);
        }
        // stage B transposed, f32 -> bf16
        #pragma unroll
        for (int t = 0; t < 4; ++t) {
            int i  = tid + 256 * t;
            int kk = i >> 5;
            int nq = (i & 31) << 2;
            float4 v = *(const float4*)(W + (size_t)(k0 + kk) * D_MODEL + n0 + nq);
            sB[(nq + 0) * 32 + kk] = f32_to_bf16(v.x);
            sB[(nq + 1) * 32 + kk] = f32_to_bf16(v.y);
            sB[(nq + 2) * 32 + kk] = f32_to_bf16(v.z);
            sB[(nq + 3) * 32 + kk] = f32_to_bf16(v.w);
        }
        wait_asynccnt0();
        __syncthreads();

        v16bf a0 = load_frag_a(sA, wrow * 32,      0, lane, 32);
        v16bf a1 = load_frag_a(sA, wrow * 32 + 16, 0, lane, 32);
        v16bf b0 = load_frag_b(sB, wcol * 32,      0, lane, 32);
        v16bf b1 = load_frag_b(sB, wcol * 32 + 16, 0, lane, 32);
        acc[0][0] = wmma_bf16(a0, b0, acc[0][0]);
        acc[0][1] = wmma_bf16(a0, b1, acc[0][1]);
        acc[1][0] = wmma_bf16(a1, b0, acc[1][0]);
        acc[1][1] = wmma_bf16(a1, b1, acc[1][1]);
        __syncthreads();
    }

    const int l  = lane & 15;
    const int hi = lane >> 4;
    #pragma unroll
    for (int i = 0; i < 2; ++i) {
        #pragma unroll
        for (int j = 0; j < 2; ++j) {
            int n = n0 + wcol * 32 + j * 16 + l;
            float bn = bias[n];
            #pragma unroll
            for (int r = 0; r < 8; ++r) {
                int m = m0 + wrow * 32 + i * 16 + r + hi * 8;
                out[(size_t)m * D_MODEL + n] = acc[i][j][r] + bn;
            }
        }
    }
}

// ---------------------------------------------------------------------------
// Host launcher
// ---------------------------------------------------------------------------
extern "C" void kernel_launch(void* const* d_in, const int* in_sizes, int n_in,
                              void* d_out, int out_size, void* d_ws, size_t ws_size,
                              hipStream_t stream) {
    const float* x  = (const float*)d_in[0];
    const float* Wq = (const float*)d_in[1];
    const float* bq = (const float*)d_in[2];
    const float* Wk = (const float*)d_in[3];
    const float* bk = (const float*)d_in[4];
    const float* Wv = (const float*)d_in[5];
    const float* bv = (const float*)d_in[6];
    const float* Wo = (const float*)d_in[7];
    const float* bo = (const float*)d_in[8];

    unsigned short* qws = (unsigned short*)d_ws;
    unsigned short* kws = qws + HEAD_ELEMS;
    unsigned short* vws = kws + HEAD_ELEMS;
    unsigned short* aws = vws + HEAD_ELEMS;

    dim3 ggrid(D_MODEL / 128, TOKENS / 64);       // (8, 64)

    gemm_qkv_kernel<<<ggrid, 256, 0, stream>>>(x, Wq, bq, qws);
    gemm_qkv_kernel<<<ggrid, 256, 0, stream>>>(x, Wk, bk, kws);
    gemm_qkv_kernel<<<ggrid, 256, 0, stream>>>(x, Wv, bv, vws);

    rope_kernel<<<(BATCH * NHEAD * SEQ * 32) / 256, 256, 0, stream>>>(qws, kws);

    flash_attn_kernel<<<dim3(SEQ / 64, NHEAD, BATCH), 128, 0, stream>>>(qws, kws, vws, aws);

    gemm_out_kernel<<<ggrid, 256, 0, stream>>>(aws, Wo, bo, (float*)d_out);
}
